// DiffusionProcess_40072044871970
// MI455X (gfx1250) — matrix operand order repaired and checked by reading, not compiled, not run on hardware
//
#include <hip/hip_runtime.h>
#include <hip/hip_bf16.h>
#include <stdint.h>

// Problem sizes (match reference)
#define B_   2048
#define R_   512
#define Z_   256
#define H_   512
#define ZR_  768          // Z + R
#define T_   16
#define BZ_  (B_ * Z_)

// bf16 weight workspace element counts (transposed [N][K] layouts)
#define W0T_N (H_ * ZR_)   // 512*768 = 393216
#define WHT_N (H_ * H_)    // 262144
#define WOT_N (Z_ * H_)    // 131072
#define TEMB_N (T_ * ZR_)  // 12288

typedef __bf16 bf16_t;
typedef __attribute__((ext_vector_type(16))) __bf16 v16bf;
typedef __attribute__((ext_vector_type(8)))  __bf16 v8bf;
typedef __attribute__((ext_vector_type(8)))  float  v8f;
typedef __attribute__((ext_vector_type(4)))  float  v4f;

// ---------------------------------------------------------------------------
// CDNA5 async global->LDS copy (ASYNCcnt-tracked) via inline asm.
// VDST = LDS byte address (low 32 bits of generic LDS pointer),
// VADDR = 64-bit global address, no SADDR ("off").
// ---------------------------------------------------------------------------
__device__ __forceinline__ void async_load_b128(const float* gsrc, float* ldst) {
    unsigned int       lds_a = (unsigned int)(uintptr_t)ldst;
    unsigned long long ga    = (unsigned long long)(uintptr_t)gsrc;
    asm volatile("global_load_async_to_lds_b128 %0, %1, off"
                 :: "v"(lds_a), "v"(ga) : "memory");
}

__device__ __forceinline__ void wait_asynccnt0() {
#if defined(__has_builtin)
#if __has_builtin(__builtin_amdgcn_s_wait_asynccnt)
    __builtin_amdgcn_s_wait_asynccnt(0);
#else
    asm volatile("s_wait_asynccnt 0" ::: "memory");
#endif
#else
    asm volatile("s_wait_asynccnt 0" ::: "memory");
#endif
}

// ---------------------------------------------------------------------------
// Prep kernel: f32->bf16 transposed weights, time embeddings, constant slabs.
// ---------------------------------------------------------------------------
__global__ __launch_bounds__(256) void prep_kernel(
    const float* __restrict__ W0, const float* __restrict__ Wh,
    const float* __restrict__ Wo, const float* __restrict__ Wt,
    const float* __restrict__ bt, const float* __restrict__ noise0,
    bf16_t* __restrict__ W0t, bf16_t* __restrict__ Wht, bf16_t* __restrict__ Wot,
    float* __restrict__ temb, float* __restrict__ out)
{
    long idx = (long)blockIdx.x * 256 + threadIdx.x;
    const float dt = 1.0f / (float)T_;
    const long e0 = W0T_N;
    const long e1 = e0 + WHT_N;
    const long e2 = e1 + WOT_N;
    const long e3 = e2 + TEMB_N;
    const long e4 = e3 + (long)(T_ + 1) * BZ_;   // sigmas slab
    const long e5 = e4 + BZ_;                    // mus[0]
    const long e6 = e5 + BZ_;                    // zs[0]

    if (idx < e0) {                              // W0t[n*ZR + k] = W0[k*H + n]
        long n = idx / ZR_, k = idx - n * ZR_;
        W0t[idx] = (bf16_t)W0[k * H_ + n];
    } else if (idx < e1) {                       // Wht[n*H + k] = Wh[k*H + n]
        long i = idx - e0;
        long n = i / H_, k = i - n * H_;
        Wht[i] = (bf16_t)Wh[k * H_ + n];
    } else if (idx < e2) {                       // Wot[n*H + k] = Wo[k*Z + n]
        long i = idx - e1;
        long n = i / H_, k = i - n * H_;
        Wot[i] = (bf16_t)Wo[k * Z_ + n];
    } else if (idx < e3) {                       // temb[t][j] = relu((t+1)*dt*Wt[j] + bt[j])
        long i = idx - e2;
        long t = i / ZR_, j = i - t * ZR_;
        float v = (float)(t + 1) * dt * Wt[j] + bt[j];
        temb[i] = v > 0.0f ? v : 0.0f;
    } else if (idx < e4) {                       // sigmas: [0]=1.0, [1..16]=sqrt(dt)=0.25
        long i = idx - e3;
        out[(long)2 * (T_ + 1) * BZ_ + i] = (i < BZ_) ? 1.0f : 0.25f;
    } else if (idx < e5) {                       // mus[0] = 0
        long i = idx - e4;
        out[(long)(T_ + 1) * BZ_ + i] = 0.0f;
    } else if (idx < e6) {                       // zs[0] = noise0
        long i = idx - e5;
        out[i] = noise0[i];
    }
}

// ---------------------------------------------------------------------------
// WMMA fragment helpers (v_wmma_f32_16x16x32_bf16 layouts per CDNA5 ISA 7.12.2)
// ---------------------------------------------------------------------------
__device__ __forceinline__ v16bf cat8(v8bf lo, v8bf hi) {
    return __builtin_shufflevector(lo, hi, 0,1,2,3,4,5,6,7,8,9,10,11,12,13,14,15);
}

// A (16x32 bf16) from LDS, row-major with element stride `stride`.
// lanes 0-15: M=lane, K {0..7, 16..23}; lanes 16-31: M=lane-16, K {8..15, 24..31}
__device__ __forceinline__ v16bf load_a(const bf16_t* X, int stride, int m0, int k0, int lane) {
    int m  = m0 + (lane & 15);
    int ko = (lane & 16) ? 8 : 0;
    const bf16_t* p = X + m * stride + k0 + ko;
    v8bf lo = *(const v8bf*)p;          // K = k0+ko .. +7
    v8bf hi = *(const v8bf*)(p + 16);   // K = k0+16+ko .. +7
    return cat8(lo, hi);
}

// B (32x16 bf16) from transposed bf16 weights Wt[n][k] (row stride Kdim).
// lanes 0-15: N=n0+lane, K {0..7, 16..23}; lanes 16-31: K {8..15, 24..31}
__device__ __forceinline__ v16bf load_b(const bf16_t* Wt, int Kdim, int n0, int k0, int lane) {
    int n = n0 + (lane & 15);
    int k = k0 + ((lane & 16) ? 8 : 0);
    const bf16_t* p = Wt + (long)n * Kdim + k;
    v8bf lo = *(const v8bf*)p;
    v8bf hi = *(const v8bf*)(p + 16);
    return cat8(lo, hi);
}

// One dense layer for one wave: rows [m0,m0+16), cols [n0, n0+16*NT).
// A from LDS (bf16), B from global bf16 transposed weights, bias f32,
// result (+bias, optional relu) stored as bf16 to LDS Hout.
template <int KDIM, int NT, bool RELU>
__device__ __forceinline__ void gemm_layer(
    const bf16_t* Xin, int xst,
    const bf16_t* __restrict__ Wt, const float* __restrict__ bias,
    bf16_t* Hout, int hst, int m0, int n0, int lane)
{
    const v8f vzero = {0.f, 0.f, 0.f, 0.f, 0.f, 0.f, 0.f, 0.f};
    v8f acc[NT];
#pragma unroll
    for (int nt = 0; nt < NT; ++nt) acc[nt] = vzero;

    for (int kt = 0; kt < KDIM / 32; ++kt) {
        v16bf a = load_a(Xin, xst, m0, kt * 32, lane);
#pragma unroll
        for (int nt = 0; nt < NT; ++nt) {
            v16bf b = load_b(Wt, KDIM, n0 + nt * 16, kt * 32, lane);
            acc[nt] = __builtin_amdgcn_wmma_f32_16x16x32_bf16(
                false, a, false, b, (short)0, acc[nt], false, false);
        }
    }

    // D layout: lane -> N = n0+nt*16+(lane&15); vgpr j -> M = m0 + j + 8*(lane>=16)
    const int nc = lane & 15;
    const int mr = m0 + ((lane >> 4) << 3);
#pragma unroll
    for (int nt = 0; nt < NT; ++nt) {
        int n = n0 + nt * 16 + nc;
        float bv = bias[n];
#pragma unroll
        for (int j = 0; j < 8; ++j) {
            float v = acc[nt][j] + bv;
            if (RELU) v = fmaxf(v, 0.0f);
            Hout[(mr + j) * hst + n] = (bf16_t)v;
        }
    }
}

// ---------------------------------------------------------------------------
// Main kernel: each block owns 32 batch rows, runs all T steps locally.
// 256 threads = 8 wave32: (mi in {0,16}) x (ng in 0..3).
// ---------------------------------------------------------------------------
__global__ __launch_bounds__(256) void diffusion_kernel(
    const float* __restrict__ r_g, const float* __restrict__ noise0,
    const float* __restrict__ noise, const float* __restrict__ b0,
    const float* __restrict__ bh, const float* __restrict__ bo,
    const bf16_t* __restrict__ W0t, const bf16_t* __restrict__ Wht,
    const bf16_t* __restrict__ Wot, const float* __restrict__ temb,
    float* __restrict__ out)
{
    constexpr int MR  = 32;        // rows per block
    constexpr int XST = ZR_ + 8;   // padded strides (keep 16B alignment, dodge bank conflicts)
    constexpr int HST = H_ + 8;

    __shared__ float  zbuf[MR][Z_];          // 32 KB  (current z, f32)
    __shared__ bf16_t x_l[MR * XST];         // 48.5 KB (layer-0 input; reused as eps f32 buf)
    __shared__ bf16_t h0_l[MR * HST];        // 32.5 KB
    __shared__ bf16_t h1_l[MR * HST];        // 32.5 KB

    const int tid   = threadIdx.x;
    const int lane  = tid & 31;
    const int w     = tid >> 5;
    const int mi    = (w & 1) * 16;          // M subtile base row
    const int ng    = w >> 1;                // N group 0..3
    const int mbase = blockIdx.x * MR;

    // ---- init: z = noise0 tile
    {
        int i  = tid >> 3;
        int c0 = (tid & 7) * (Z_ / 8);       // 32 floats per thread
#pragma unroll
        for (int c = 0; c < 32; c += 4) {
            *(v4f*)&zbuf[i][c0 + c] = *(const v4f*)(noise0 + (size_t)(mbase + i) * Z_ + c0 + c);
        }
    }
    __syncthreads();

    const float dt = 1.0f / (float)T_;
    const float sd = 0.25f;                  // sqrt(dt)

    for (int t = 0; t < T_; ++t) {
        // ---- Phase A: x = concat(z, r) + temb[t]  (bf16 into LDS)
        {
            const float* tb = temb + t * ZR_;
            int i  = tid >> 3;
            int c0 = (tid & 7) * (ZR_ / 8);  // 96 cols per thread
            for (int c = c0; c < c0 + 96; c += 4) {
                v4f v;
                if (c < Z_) v = *(const v4f*)&zbuf[i][c];
                else        v = *(const v4f*)(r_g + (size_t)(mbase + i) * R_ + (c - Z_));
                v4f tv = *(const v4f*)(tb + c);
                bf16_t* d = &x_l[i * XST + c];
                d[0] = (bf16_t)(v.x + tv.x);
                d[1] = (bf16_t)(v.y + tv.y);
                d[2] = (bf16_t)(v.z + tv.z);
                d[3] = (bf16_t)(v.w + tv.w);
            }
        }
        __syncthreads();

        // ---- Layer 0: h0 = relu(x @ W0 + b0)   (K=768, N=512)
        gemm_layer<ZR_, 8, true>(x_l, XST, W0t, b0, h0_l, HST, mi, ng * 128, lane);
        __syncthreads();

        // ---- async prefetch: eps = noise[t] tile into LDS (x_l space is free now)
        {
            float* epsf = (float*)x_l;
            const float* esrc = noise + (size_t)t * BZ_ + (size_t)mbase * Z_;
#pragma unroll
            for (int q = 0; q < 8; ++q) {
                int off = (tid + q * 256) * 4;   // 4 floats (16B) per request, contiguous
                async_load_b128(esrc + off, epsf + off);
            }
        }

        // ---- Layer 1: h1 = h0 @ Wh + bh        (no relu between shared layers)
        gemm_layer<H_, 8, false>(h0_l, HST, Wht, bh, h1_l, HST, mi, ng * 128, lane);
        __syncthreads();

        // ---- Layer 2: h0 = h1 @ Wh + bh
        gemm_layer<H_, 8, false>(h1_l, HST, Wht, bh, h0_l, HST, mi, ng * 128, lane);
        __syncthreads();

        // ---- Layer 3: s = h0 @ Wo + bo  (K=512, N=256), then Euler update
        {
            const v8f vzero = {0.f, 0.f, 0.f, 0.f, 0.f, 0.f, 0.f, 0.f};
            v8f acc[4];
#pragma unroll
            for (int nt = 0; nt < 4; ++nt) acc[nt] = vzero;

            for (int kt = 0; kt < H_ / 32; ++kt) {
                v16bf a = load_a(h0_l, HST, mi, kt * 32, lane);
#pragma unroll
                for (int nt = 0; nt < 4; ++nt) {
                    v16bf b = load_b(Wot, H_, ng * 64 + nt * 16, kt * 32, lane);
                    acc[nt] = __builtin_amdgcn_wmma_f32_16x16x32_bf16(
                        false, a, false, b, (short)0, acc[nt], false, false);
                }
            }

            wait_asynccnt0();                       // own async eps writes complete
            __syncthreads();                        // all waves' eps visible

            const int nc = lane & 15;
            const int mr = mi + ((lane >> 4) << 3);
            const float* epsf = (const float*)x_l;
#pragma unroll
            for (int nt = 0; nt < 4; ++nt) {
                int n = ng * 64 + nt * 16 + nc;
                float bov = bo[n];
#pragma unroll
                for (int j = 0; j < 8; ++j) {
                    int m = mr + j;
                    float s    = acc[nt][j] + bov;
                    float zold = zbuf[m][n];
                    float mu   = fmaf(s, dt, zold);
                    float e    = epsf[m * Z_ + n];
                    float zn   = fmaf(sd, e, mu);
                    zbuf[m][n] = zn;
                    size_t gi = (size_t)(mbase + m) * Z_ + n;
                    out[(size_t)(t + 1) * BZ_ + gi]          = zn;  // zs slab
                    out[(size_t)(T_ + 1 + t + 1) * BZ_ + gi] = mu;  // mus slab
                }
            }
        }
        __syncthreads();
    }
}

// ---------------------------------------------------------------------------
extern "C" void kernel_launch(void* const* d_in, const int* in_sizes, int n_in,
                              void* d_out, int out_size, void* d_ws, size_t ws_size,
                              hipStream_t stream)
{
    (void)in_sizes; (void)n_in; (void)out_size; (void)ws_size;

    const float* r      = (const float*)d_in[0];
    const float* noise0 = (const float*)d_in[1];
    const float* noise  = (const float*)d_in[2];
    const float* W0     = (const float*)d_in[3];
    const float* b0     = (const float*)d_in[4];
    const float* Wh     = (const float*)d_in[5];
    const float* bh     = (const float*)d_in[6];
    const float* Wo     = (const float*)d_in[7];
    const float* bo     = (const float*)d_in[8];
    const float* Wt     = (const float*)d_in[9];
    const float* bt     = (const float*)d_in[10];
    float* out = (float*)d_out;

    bf16_t* W0t  = (bf16_t*)d_ws;
    bf16_t* Wht  = W0t + W0T_N;
    bf16_t* Wot  = Wht + WHT_N;
    float*  temb = (float*)(Wot + WOT_N);    // byte offset 1572864, 16B aligned

    const long total = (long)W0T_N + WHT_N + WOT_N + TEMB_N +
                       (long)(T_ + 1) * BZ_ + BZ_ + BZ_;
    const int pblocks = (int)((total + 255) / 256);

    prep_kernel<<<pblocks, 256, 0, stream>>>(W0, Wh, Wo, Wt, bt, noise0,
                                             W0t, Wht, Wot, temb, out);
    diffusion_kernel<<<B_ / 32, 256, 0, stream>>>(r, noise0, noise, b0, bh, bo,
                                                  W0t, Wht, Wot, temb, out);
}
